// DecoderNetwork_9466107921169
// MI455X (gfx1250) — matrix operand order repaired and checked by reading, not compiled, not run on hardware
//
#include <hip/hip_runtime.h>
#include <hip/hip_bf16.h>
#include <math.h>

// ---------------------------------------------------------------------------
// Types
// ---------------------------------------------------------------------------
typedef _Float16 half_t;
typedef __attribute__((ext_vector_type(16))) _Float16 v16h;
typedef __attribute__((ext_vector_type(8)))  _Float16 v8h;
typedef __attribute__((ext_vector_type(8)))  float    v8f;

#define EPSF   1e-5f
#define SLOPEF 0.01f

// ---------------------------------------------------------------------------
// Zero scratch (stats + padded f16 activation buffers must start at 0)
// ---------------------------------------------------------------------------
__global__ void zero_u32(unsigned int* __restrict__ p, long n) {
  long i = (long)blockIdx.x * 256 + threadIdx.x;
  if (i < n) p[i] = 0u;
}

// ---------------------------------------------------------------------------
// Weight repack: raw f32 (taps, Cin, Cout) -> f16 WMMA B-fragment layout
//   frag[(((t*KC + kc)*NT + nt)*512) + lane*16 + e]
// B (K x N) per-lane layout (mirrors sparse-B ISA table at K=32):
//   N = lane & 15 ; K = kc*32 + ((lane & 16) ? 8 : 0) + e + ((e & 8) ? 8 : 0)
// grid = (ntaps, KC, NT), block = 512 (= lane*16 + e)  -> no divisions
// ---------------------------------------------------------------------------
__global__ __launch_bounds__(512)
void build_frag(const float* __restrict__ w, half_t* __restrict__ frag,
                int Cin, int Cout) {
  const int t  = blockIdx.x;
  const int kc = blockIdx.y;
  const int nt = blockIdx.z;
  const int KC = gridDim.y, NT = gridDim.z;
  const int tid  = threadIdx.x;       // 0..511
  const int e    = tid & 15;
  const int lane = tid >> 4;          // 0..31
  const int n = nt * 16 + (lane & 15);
  const int k = kc * 32 + ((lane & 16) ? 8 : 0) + e + ((e & 8) ? 8 : 0);
  float v = 0.f;
  if (k < Cin && n < Cout) v = w[((long)t * Cin + k) * Cout + n];
  frag[(((long)t * KC + kc) * NT + nt) * 512 + tid] = (half_t)v;
}

// ---------------------------------------------------------------------------
// Generic N-D conv as implicit GEMM with WMMA f16.
// One wave computes a 16(M) x (16*NTL)(N) output tile: the A fragment is
// loaded once per K-chunk and reused by NTL back-to-back WMMAs.
// All dims are powers of two -> pure shift/mask addressing (no int division).
//   act: f16 input, layout (pos_src << lgKpad) with pos over (B, S>>upShift..)
//   out: f32 (pos, Cout); stats: per-channel {sum[64], sumsq[64]} via atomics
// ---------------------------------------------------------------------------
struct ConvDims {
  int lgS[5];    // log2 of output spatial grid per dim
  int padlo[5];  // low zero-padding per dim
  int upShift;   // 1 -> source activations are at half resolution (fused up2)
  int lgKpad;    // log2 padded input channels (5/6/7)
  int KC;        // Kpad / 32
  int Cout;      // real output channels
};

template <int ND> struct Pow3 { static constexpr int v = 3 * Pow3<ND - 1>::v; };
template <> struct Pow3<0>    { static constexpr int v = 1; };

template <int ND, int NTL>
__global__ __launch_bounds__(32)
void conv_wmma(const half_t* __restrict__ act, const half_t* __restrict__ frag,
               float* __restrict__ outRaw, float* __restrict__ stats, ConvDims cd) {
  constexpr int NTAPS = Pow3<ND>::v;
  const int lane = threadIdx.x;
  const unsigned gm = blockIdx.x * 16u + (lane & 15);

  // decompose output position -> (batch, x[0..ND-1]); x[ND-1] fastest
  int x[ND];
  unsigned rem = gm;
#pragma unroll
  for (int i = ND - 1; i >= 0; --i) {
    x[i] = (int)(rem & ((1u << cd.lgS[i]) - 1u));
    rem >>= cd.lgS[i];
  }
  const unsigned bIdx = rem;

  const v8f vzero = {};
  v8f acc[NTL];
#pragma unroll
  for (int q = 0; q < NTL; ++q) acc[q] = vzero;

  const int KC     = cd.KC;
  const int k0base = (lane & 16) ? 8 : 0;  // A-layout: half-wave K offset

  int ti[ND];  // base-3 tap odometer (avoids %3 per tap)
#pragma unroll
  for (int i = 0; i < ND; ++i) ti[i] = 0;

#pragma clang loop unroll(disable)
  for (int t = 0; t < NTAPS; ++t) {
    bool valid = true;
    unsigned off = bIdx;
#pragma unroll
    for (int i = 0; i < ND; ++i) {
      int xi = x[i] + ti[i] - cd.padlo[i];
      if ((unsigned)xi >= (1u << cd.lgS[i])) valid = false;  // zero padding
      off = (off << (cd.lgS[i] - cd.upShift)) + (unsigned)(xi >> cd.upShift);
    }
    const half_t* ap = act + ((size_t)off << cd.lgKpad);

#pragma clang loop unroll(disable)
    for (int kc = 0; kc < KC; ++kc) {
      // A 16x32 f16 layout: lanes 0-15 hold K {0-7,16-23}, lanes 16-31 {8-15,24-31}
      v8h lo = {}, hi = {};
      if (valid) {
        const half_t* p0 = ap + kc * 32 + k0base;
        lo = *(const v8h*)(p0);
        hi = *(const v8h*)(p0 + 16);
      }
      v16h a = __builtin_shufflevector(lo, hi, 0, 1, 2, 3, 4, 5, 6, 7,
                                               8, 9, 10, 11, 12, 13, 14, 15);
      const half_t* fb = frag + (((long)(t * KC + kc) * NTL) << 9) + lane * 16;
#pragma unroll
      for (int q = 0; q < NTL; ++q) {
        v16h b = *(const v16h*)(fb + (q << 9));
        acc[q] = __builtin_amdgcn_wmma_f32_16x16x32_f16(false, a, false, b,
                                                        (short)0, acc[q], false, false);
      }
    }

    // odometer increment (fastest dim last)
#pragma unroll
    for (int i = ND - 1; i >= 0; --i) {
      if (++ti[i] < 3) break;
      ti[i] = 0;
    }
  }

  // C/D layout: VGPR r: lanes 0-15 -> (M=r, N=lane), lanes 16-31 -> (M=r+8, N=lane-16)
  const unsigned mbase = blockIdx.x * 16u + ((lane & 16) ? 8u : 0u);
#pragma unroll
  for (int q = 0; q < NTL; ++q) {
    const int n = (lane & 15) + q * 16;
    if (n < cd.Cout) {
      float s = 0.f, ss = 0.f;
#pragma unroll
      for (int r = 0; r < 8; ++r) {
        float v = acc[q][r];
        outRaw[(size_t)(mbase + r) * cd.Cout + n] = v;
        s += v; ss += v * v;
      }
      atomicAdd(&stats[n], s);
      atomicAdd(&stats[64 + n], ss);
    }
  }
}

// ---------------------------------------------------------------------------
// BN (train-mode, batch stats) + leaky-ReLU + optional nearest-2x upsample +
// reshape (contiguous identity) + f32->f16 pack into next layer's padded act.
// grid = (ceil(Pout/256), C): channel from blockIdx.y -> no divisions.
// Reshape: lgC >= 0 (C pow2, shift-based) else C == Cnext (identity).
// ---------------------------------------------------------------------------
struct ApplyDims {
  int   lgS[5];     // log2 conv output grid (pre-upsample)
  int   nd;
  int   C;          // conv Cout
  int   up;         // 1 -> 2x upsample every spatial dim
  int   lgC;        // log2(C) or -1 (then Cnext == C)
  int   lgCnext;    // log2 next conv's real Cin (valid iff lgC >= 0)
  int   lgKpadNext; // log2 next conv's padded Cin
  unsigned Pout;    // #positions written
  float invM;       // 1 / (#positions of conv output)
};

__global__ __launch_bounds__(256)
void bn_act_pack(const float* __restrict__ raw, const float* __restrict__ stats,
                 const float* __restrict__ sc, const float* __restrict__ bi,
                 half_t* __restrict__ actOut, ApplyDims ad) {
  unsigned p = blockIdx.x * 256u + threadIdx.x;
  if (p >= ad.Pout) return;
  const int c = blockIdx.y;
  unsigned pSrc = p;
  if (ad.up) {
    int xo[5];
    unsigned rem = p;
    for (int i = ad.nd - 1; i >= 0; --i) {
      int lg = ad.lgS[i] + 1;                 // upsampled dim
      xo[i] = (int)(rem & ((1u << lg) - 1u));
      rem >>= lg;
    }
    unsigned src = rem;                       // batch
    for (int i = 0; i < ad.nd; ++i) src = (src << ad.lgS[i]) | (unsigned)(xo[i] >> 1);
    pSrc = src;
  }
  float mean = stats[c] * ad.invM;
  float var  = stats[64 + c] * ad.invM - mean * mean;
  float y = (raw[(size_t)pSrc * ad.C + c] - mean) * rsqrtf(var + EPSF) * sc[c] + bi[c];
  y = y > 0.f ? y : SLOPEF * y;
  unsigned pN; int cN;
  if (ad.lgC >= 0) {
    unsigned idx = (p << ad.lgC) | (unsigned)c; // logical contiguous element id
    pN = idx >> ad.lgCnext;
    cN = (int)(idx & ((1u << ad.lgCnext) - 1u));
  } else {                                      // C == Cnext (24): identity
    pN = p; cN = c;
  }
  actOut[((size_t)pN << ad.lgKpadNext) + cN] = (half_t)y;
}

// ---------------------------------------------------------------------------
// MLP: three Dense(no bias)+BN+leakyReLU stages in one 64-thread block.
// ---------------------------------------------------------------------------
__device__ void bn_lrelu_batch(float* h, int n, float* sh, float* mv,
                               const float* sc, const float* bi) {
  const int tb = threadIdx.x;
  for (int j = 0; j < n; ++j) sh[j * 64 + tb] = h[j];
  __syncthreads();
  if (tb < n) {
    float s = 0.f, ss = 0.f;
    for (int i = 0; i < 64; ++i) { float v = sh[tb * 64 + i]; s += v; ss += v * v; }
    float m   = s * (1.f / 64.f);
    float var = ss * (1.f / 64.f) - m * m;
    mv[tb]      = m;
    mv[64 + tb] = rsqrtf(var + EPSF);
  }
  __syncthreads();
  for (int j = 0; j < n; ++j) {
    float y = (h[j] - mv[j]) * mv[64 + j] * sc[j] + bi[j];
    h[j] = y > 0.f ? y : SLOPEF * y;
  }
  __syncthreads();
}

__global__ __launch_bounds__(64)
void mlp_kernel(const float* __restrict__ inp,
                const float* __restrict__ W0, const float* __restrict__ W1,
                const float* __restrict__ W2,
                const float* __restrict__ s0, const float* __restrict__ b0,
                const float* __restrict__ s1, const float* __restrict__ b1,
                const float* __restrict__ s2, const float* __restrict__ b2,
                half_t* __restrict__ act) {
  __shared__ float sh[64 * 64];
  __shared__ float mv[128];
  const int tb = threadIdx.x;  // batch row (B = 64)
  float h[64];
  float x0 = inp[tb * 2 + 0], x1 = inp[tb * 2 + 1];
  h[0] = x0 * W0[0] + x1 * W0[2];
  h[1] = x0 * W0[1] + x1 * W0[3];
  bn_lrelu_batch(h, 2, sh, mv, s0, b0);
  float g[8];
  for (int j = 0; j < 8; ++j) g[j] = h[0] * W1[j] + h[1] * W1[8 + j];
  bn_lrelu_batch(g, 8, sh, mv, s1, b1);
  for (int j = 0; j < 64; ++j) {
    float s = 0.f;
    for (int k = 0; k < 8; ++k) s += g[k] * W2[k * 64 + j];
    h[j] = s;
  }
  bn_lrelu_batch(h, 64, sh, mv, s2, b2);
  // (64,64) -> (64,2,2,16) contiguous; write padded-32 f16 act for conv2a
  for (int j = 0; j < 64; ++j) {
    int p = tb * 4 + (j >> 4);
    int c = j & 15;
    act[p * 32 + c] = (half_t)h[j];
  }
}

// ---------------------------------------------------------------------------
// Final: BN (train stats) + softmax over 4 channels -> d_out (f32)
// ---------------------------------------------------------------------------
__global__ __launch_bounds__(256)
void bn_softmax4(const float* __restrict__ raw, const float* __restrict__ stats,
                 const float* __restrict__ sc, const float* __restrict__ bi,
                 float* __restrict__ out, unsigned P, float invM) {
  unsigned p = blockIdx.x * 256u + threadIdx.x;
  if (p >= P) return;
  float v[4];
  float mx = -3.4e38f;
#pragma unroll
  for (int c = 0; c < 4; ++c) {
    float mean = stats[c] * invM;
    float var  = stats[64 + c] * invM - mean * mean;
    float y = (raw[(size_t)p * 4 + c] - mean) * rsqrtf(var + EPSF) * sc[c] + bi[c];
    v[c] = y;
    mx = fmaxf(mx, y);
  }
  float s = 0.f;
#pragma unroll
  for (int c = 0; c < 4; ++c) { v[c] = __expf(v[c] - mx); s += v[c]; }
  float inv = 1.0f / s;
#pragma unroll
  for (int c = 0; c < 4; ++c) out[(size_t)p * 4 + c] = v[c] * inv;
}

// ---------------------------------------------------------------------------
// Host orchestration
// ---------------------------------------------------------------------------
extern "C" void kernel_launch(void* const* d_in, const int* in_sizes, int n_in,
                              void* d_out, int out_size, void* d_ws, size_t ws_size,
                              hipStream_t stream) {
  (void)in_sizes; (void)n_in; (void)out_size; (void)ws_size;
  const float* inp = (const float*)d_in[0];
  const float* W[13];
  for (int i = 0; i < 13; ++i) W[i] = (const float*)d_in[1 + i];
  // BN scale/bias: order m0,m1,m2,c2a,c2b,c2c,c3a,c3b,c4a,c4b,c5a,c5b,out
  const float *Sg[13], *Bg[13];
  for (int k = 0; k < 13; ++k) { Sg[k] = (const float*)d_in[14 + 2 * k]; Bg[k] = (const float*)d_in[15 + 2 * k]; }

  size_t off = 0;
  auto alloc = [&](size_t bytes) -> void* {
    off = (off + 255) & ~(size_t)255;
    void* p = (char*)d_ws + off;
    off += bytes;
    return p;
  };

  // stats (10 conv layers x {sum[64], sumsq[64]}) then padded f16 activations:
  float* stats = (float*)alloc(10 * 128 * sizeof(float));
  half_t* a2a = (half_t*)alloc((size_t)256 * 32 * 2);
  half_t* a2b = (half_t*)alloc((size_t)1024 * 32 * 2);
  half_t* a2c = (half_t*)alloc((size_t)1024 * 32 * 2);
  half_t* a3a = (half_t*)alloc((size_t)4096 * 32 * 2);
  half_t* a3b = (half_t*)alloc((size_t)4096 * 32 * 2);
  half_t* a4a = (half_t*)alloc((size_t)16384 * 64 * 2);
  half_t* a4b = (half_t*)alloc((size_t)16384 * 32 * 2);
  half_t* a5a = (half_t*)alloc((size_t)65536 * 128 * 2);
  half_t* a5b = (half_t*)alloc((size_t)65536 * 32 * 2);
  half_t* aot = (half_t*)alloc((size_t)65536 * 32 * 2);
  size_t zeroEnd = off;  // everything above must start zeroed (pad regions)
  // weight fragments
  half_t* f2a = (half_t*)alloc((size_t)9 * 1 * 1 * 512 * 2);
  half_t* f2b = (half_t*)alloc((size_t)9 * 1 * 1 * 512 * 2);
  half_t* f2c = (half_t*)alloc((size_t)9 * 1 * 1 * 512 * 2);
  half_t* f3a = (half_t*)alloc((size_t)27 * 1 * 2 * 512 * 2);
  half_t* f3b = (half_t*)alloc((size_t)27 * 1 * 2 * 512 * 2);
  half_t* f4a = (half_t*)alloc((size_t)81 * 2 * 2 * 512 * 2);
  half_t* f4b = (half_t*)alloc((size_t)81 * 1 * 2 * 512 * 2);
  half_t* f5a = (half_t*)alloc((size_t)243 * 4 * 2 * 512 * 2);
  half_t* f5b = (half_t*)alloc((size_t)243 * 1 * 2 * 512 * 2);
  half_t* fot = (half_t*)alloc((size_t)243 * 1 * 1 * 512 * 2);
  // raw f32 conv outputs
  float* r2a = (float*)alloc((size_t)256 * 16 * 4);
  float* r2b = (float*)alloc((size_t)1024 * 16 * 4);
  float* r2c = (float*)alloc((size_t)1024 * 16 * 4);
  float* r3a = (float*)alloc((size_t)4096 * 32 * 4);
  float* r3b = (float*)alloc((size_t)4096 * 32 * 4);
  float* r4a = (float*)alloc((size_t)16384 * 32 * 4);
  float* r4b = (float*)alloc((size_t)16384 * 32 * 4);
  float* r5a = (float*)alloc((size_t)65536 * 24 * 4);
  float* r5b = (float*)alloc((size_t)65536 * 24 * 4);
  float* rot = (float*)alloc((size_t)2097152 * 4 * 4);

  // 1) zero stats + padded activations
  long zw = (long)(zeroEnd / 4);
  zero_u32<<<dim3((unsigned)((zw + 255) / 256)), dim3(256), 0, stream>>>((unsigned int*)d_ws, zw);

  // 2) repack weights into WMMA fragments: grid (taps, KC, NT) x 512 threads
  auto buildFrag = [&](const float* w, half_t* frag, int ntaps, int Cin, int KC, int Cout, int NT) {
    build_frag<<<dim3((unsigned)ntaps, (unsigned)KC, (unsigned)NT), dim3(512), 0, stream>>>(
        w, frag, Cin, Cout);
  };
  buildFrag(W[3],  f2a, 9,   16,  1, 16, 1);
  buildFrag(W[4],  f2b, 9,   16,  1, 16, 1);
  buildFrag(W[5],  f2c, 9,   16,  1, 16, 1);
  buildFrag(W[6],  f3a, 27,  16,  1, 32, 2);
  buildFrag(W[7],  f3b, 27,  32,  1, 32, 2);
  buildFrag(W[8],  f4a, 81,  64,  2, 32, 2);
  buildFrag(W[9],  f4b, 81,  32,  1, 32, 2);
  buildFrag(W[10], f5a, 243, 128, 4, 24, 2);
  buildFrag(W[11], f5b, 243, 24,  1, 24, 2);
  buildFrag(W[12], fot, 243, 24,  1, 4,  1);

  // 3) MLP head
  mlp_kernel<<<dim3(1), dim3(64), 0, stream>>>(inp, W[0], W[1], W[2],
      Sg[0], Bg[0], Sg[1], Bg[1], Sg[2], Bg[2], a2a);

  // 4) conv pipeline (one wave = 16 x (16*NT) tile)
  auto runConv = [&](int nd, const int* lgS, const int* padlo, int upShift,
                     int lgKpad, int KC, int Cout, int NT, long M,
                     const half_t* actIn, const half_t* frag, float* raw, float* st) {
    ConvDims cd{};
    for (int i = 0; i < 5; ++i) { cd.lgS[i] = (i < nd) ? lgS[i] : 0; cd.padlo[i] = (i < nd) ? padlo[i] : 0; }
    cd.upShift = upShift; cd.lgKpad = lgKpad; cd.KC = KC; cd.Cout = Cout;
    dim3 grid((unsigned)(M / 16));
    switch (nd * 10 + NT) {
      case 21: conv_wmma<2, 1><<<grid, dim3(32), 0, stream>>>(actIn, frag, raw, st, cd); break;
      case 32: conv_wmma<3, 2><<<grid, dim3(32), 0, stream>>>(actIn, frag, raw, st, cd); break;
      case 42: conv_wmma<4, 2><<<grid, dim3(32), 0, stream>>>(actIn, frag, raw, st, cd); break;
      case 52: conv_wmma<5, 2><<<grid, dim3(32), 0, stream>>>(actIn, frag, raw, st, cd); break;
      case 51: conv_wmma<5, 1><<<grid, dim3(32), 0, stream>>>(actIn, frag, raw, st, cd); break;
    }
  };
  auto runApply = [&](int nd, const int* lgS, int C, int lgC, int up, long M,
                      const float* raw, const float* st, const float* sc, const float* bi,
                      int lgCnext, int lgKpadNext, half_t* actOut) {
    ApplyDims ad{};
    for (int i = 0; i < 5; ++i) ad.lgS[i] = (i < nd) ? lgS[i] : 0;
    ad.nd = nd; ad.C = C; ad.up = up; ad.lgC = lgC; ad.lgCnext = lgCnext;
    ad.lgKpadNext = lgKpadNext;
    long Pout = up ? (M << nd) : M;
    ad.Pout = (unsigned)Pout;
    ad.invM = 1.0f / (float)M;
    dim3 grid((unsigned)((Pout + 255) / 256), (unsigned)C);
    bn_act_pack<<<grid, dim3(256), 0, stream>>>(raw, st, sc, bi, actOut, ad);
  };

  const int L2a[2] = {1, 1}, L2b[2] = {2, 2}, P2[2] = {1, 1};
  const int L3[3] = {2, 2, 2}, P3[3] = {1, 1, 1};
  const int L4[4] = {2, 2, 2, 2}, P4[4] = {0, 1, 1, 1};
  const int L5[5] = {2, 2, 2, 2, 2}, P5[5] = {0, 0, 1, 1, 1};
  const int L5o[5] = {3, 3, 3, 3, 3};

  runConv(2, L2a, P2, 0, 5, 1, 16, 1, 256, a2a, f2a, r2a, stats + 0 * 128);
  runApply(2, L2a, 16, 4, 1, 256, r2a, stats + 0 * 128, Sg[3], Bg[3], 4, 5, a2b);

  runConv(2, L2b, P2, 0, 5, 1, 16, 1, 1024, a2b, f2b, r2b, stats + 1 * 128);
  runApply(2, L2b, 16, 4, 0, 1024, r2b, stats + 1 * 128, Sg[4], Bg[4], 4, 5, a2c);

  runConv(2, L2b, P2, 0, 5, 1, 16, 1, 1024, a2c, f2c, r2c, stats + 2 * 128);
  runApply(2, L2b, 16, 4, 1, 1024, r2c, stats + 2 * 128, Sg[5], Bg[5], 4, 5, a3a);

  runConv(3, L3, P3, 0, 5, 1, 32, 2, 4096, a3a, f3a, r3a, stats + 3 * 128);
  runApply(3, L3, 32, 5, 0, 4096, r3a, stats + 3 * 128, Sg[6], Bg[6], 5, 5, a3b);

  runConv(3, L3, P3, 0, 5, 1, 32, 2, 4096, a3b, f3b, r3b, stats + 4 * 128);
  runApply(3, L3, 32, 5, 1, 4096, r3b, stats + 4 * 128, Sg[7], Bg[7], 6, 6, a4a);

  runConv(4, L4, P4, 0, 6, 2, 32, 2, 16384, a4a, f4a, r4a, stats + 5 * 128);
  runApply(4, L4, 32, 5, 0, 16384, r4a, stats + 5 * 128, Sg[8], Bg[8], 5, 5, a4b);

  runConv(4, L4, P4, 0, 5, 1, 32, 2, 16384, a4b, f4b, r4b, stats + 6 * 128);
  runApply(4, L4, 32, 5, 1, 16384, r4b, stats + 6 * 128, Sg[9], Bg[9], 7, 7, a5a);

  runConv(5, L5, P5, 0, 7, 4, 24, 2, 65536, a5a, f5a, r5a, stats + 7 * 128);
  runApply(5, L5, 24, -1, 0, 65536, r5a, stats + 7 * 128, Sg[10], Bg[10], 0, 5, a5b);

  runConv(5, L5, P5, 0, 5, 1, 24, 2, 65536, a5b, f5b, r5b, stats + 8 * 128);
  runApply(5, L5, 24, -1, 0, 65536, r5b, stats + 8 * 128, Sg[11], Bg[11], 0, 5, aot);

  // output conv on the 8^5 grid, reading 4^5-grid activations (fused up2)
  runConv(5, L5o, P5, 1, 5, 1, 4, 1, 2097152, aot, fot, rot, stats + 9 * 128);

  // 5) BN + softmax -> d_out
  bn_softmax4<<<dim3((unsigned)((2097152 + 255) / 256)), dim3(256), 0, stream>>>(
      rot, stats + 9 * 128, Sg[12], Bg[12], (float*)d_out, 2097152u,
      1.0f / 2097152.0f);
}
